// SAM_48507360641092
// MI455X (gfx1250) — compile-verified, matrix operand order checked by reading, not run on hardware
//
#include <hip/hip_runtime.h>
#include <hip/hip_bf16.h>

typedef __attribute__((ext_vector_type(16))) _Float16 v16h;
typedef __attribute__((ext_vector_type(8)))  float    v8f;
typedef __attribute__((ext_vector_type(4)))  float    f32x4;
typedef __attribute__((ext_vector_type(4)))  int      v4i;

constexpr int BS = 8;
constexpr int C  = 64;     // channels
constexpr int D  = 32;     // c//2 (q/k dim)
constexpr int HW = 4096;   // 64*64 spatial
constexpr int MT = 64;     // keys processed per iteration
constexpr int WAVES = 8;   // waves per workgroup (128 queries/WG -> 2x K/V reuse)
constexpr int KLD = 40;    // LDS pitch (halves) for K tile (32 + 8 pad)
constexpr int VLD = 72;    // LDS pitch (halves) for V tile (64 + 8 pad)

// gfx1250 async global->LDS DMA (ASYNCcnt-tracked); falls back to a
// software-pipelined register staging path if the builtin is absent.
#if defined(__has_builtin)
#  if __has_builtin(__builtin_amdgcn_global_load_async_to_lds_b128)
#    define USE_ASYNC_LDS 1
#  endif
#endif

#if defined(USE_ASYNC_LDS)
__device__ inline void async_b128(const _Float16* g, _Float16* l) {
    __builtin_amdgcn_global_load_async_to_lds_b128(
        (__attribute__((address_space(1))) v4i*)(g),
        (__attribute__((address_space(3))) v4i*)(l), 0, 0);
}
#endif

union FragU {
    v16h h;
    struct { f32x4 a, b; } q;
    _Float16 e[16];
};

// Load a 16-lane x 16-half WMMA A/B fragment per the CDNA5 ISA 16-bit layout:
// lane L holds row/col (L&15); halves i=0..7  -> k = (L>>4)*8 + i,
//                              halves i=8..15 -> k = 16 + (L>>4)*8 + (i-8).
// Requires ld % 8 == 0 so each chunk is 16-byte aligned (-> b128 loads).
template <typename P>
__device__ inline v16h load_frag(const P* base, int ld, int lane) {
    const int row = lane & 15;
    const int sel = lane >> 4;
    const _Float16* p = (const _Float16*)base + (size_t)row * ld + sel * 8;
    FragU u;
    u.q.a = *(const f32x4*)(p);        // k = sel*8 + 0..7
    u.q.b = *(const f32x4*)(p + 16);   // k = 16 + sel*8 + 0..7
    return u.h;
}

// ---------------------------------------------------------------------------
// Kernel 1: 1x1-conv projections.  Q is PRE-SCALED by log2(e) so the
// attention softmax runs natively in the exp2 domain (saves one VALU mul per
// energy element; softmax is shift/scale-of-log-base invariant).
//   Q (f16 [b][n][d]), K^T (f16 [b][m][d]), V (f16 [b][c][m]).
// ---------------------------------------------------------------------------
__global__ __launch_bounds__(256) void proj_kernel(
    const float* __restrict__ x,  const float* __restrict__ wq,
    const float* __restrict__ wk, const float* __restrict__ wv,
    _Float16* __restrict__ Qh, _Float16* __restrict__ Kt,
    _Float16* __restrict__ Vh)
{
    __shared__ float sW[(D + D + C) * C];  // 8192 floats = 32 KB
    for (int i = threadIdx.x; i < D * C; i += 256) sW[i] = wq[i];
    for (int i = threadIdx.x; i < D * C; i += 256) sW[D * C + i] = wk[i];
    for (int i = threadIdx.x; i < C * C; i += 256) sW[2 * D * C + i] = wv[i];
    __syncthreads();

    const int gid = blockIdx.x * 256 + threadIdx.x;
    const int b = gid >> 12;        // / 4096
    const int n = gid & (HW - 1);

    const float* xb = x + ((size_t)b * C) * HW + n;
    float xv[C];
#pragma unroll
    for (int c = 0; c < C; ++c) xv[c] = xb[(size_t)c * HW];

    constexpr float LOG2E = 1.4426950408889634f;
    _Float16* qp = Qh + ((size_t)b * HW + n) * D;
    _Float16* kp = Kt + ((size_t)b * HW + n) * D;
#pragma unroll 4
    for (int d = 0; d < D; ++d) {
        float aq = 0.f, ak = 0.f;
#pragma unroll
        for (int c = 0; c < C; ++c) {
            aq += sW[d * C + c] * xv[c];
            ak += sW[D * C + d * C + c] * xv[c];
        }
        qp[d] = (_Float16)(aq * LOG2E);   // exp2-domain Q
        kp[d] = (_Float16)ak;
    }

    _Float16* vp = Vh + (size_t)b * C * HW + n;
#pragma unroll 4
    for (int d = 0; d < C; ++d) {
        float av = 0.f;
#pragma unroll
        for (int c = 0; c < C; ++c) av += sW[2 * D * C + d * C + c] * xv[c];
        vp[(size_t)d * HW] = (_Float16)av;
    }
}

// ---------------------------------------------------------------------------
// Kernel 2: fused flash attention, fully transposed formulation.
//   S^T = K * Q^T   -> lane holds ONE query column n (scalar softmax stats)
//   O^T = V * P^T   (P^T built in registers from S^T D-layout, no LDS)
// grid = (HW/128, BS), block = 256 (8 wave32s); each wave owns 16 queries.
// K/V tiles double-buffered in LDS, staged by async DMA when available.
// ---------------------------------------------------------------------------
__global__ __launch_bounds__(256) void attn_kernel(
    const _Float16* __restrict__ Qh, const _Float16* __restrict__ Kt,
    const _Float16* __restrict__ Vh, float* __restrict__ out)
{
#if defined(USE_ASYNC_LDS)
    __shared__ _Float16 sK[2][MT * KLD];   // K^T tile: [m][d]
    __shared__ _Float16 sV[2][C * VLD];    // V tile:   [c][m]
#else
    __shared__ _Float16 sK[1][MT * KLD];
    __shared__ _Float16 sV[1][C * VLD];
#endif

    const int tid  = threadIdx.x;
    const int wave = tid >> 5;
    const int lane = tid & 31;
    const int b    = blockIdx.y;
    const int n0   = blockIdx.x * (WAVES * 16) + wave * 16;

    const _Float16* Qb = Qh + ((size_t)b * HW) * D;
    const _Float16* Kb = Kt + ((size_t)b * HW) * D;
    const _Float16* Vb = Vh + ((size_t)b * C) * HW;

    // Q^T fragment (B operand of GEMM1) lives in registers for the whole pass.
    const v16h qf = load_frag(Qb + (size_t)n0 * D, D, lane);

    const v8f vzero = {0.f, 0.f, 0.f, 0.f, 0.f, 0.f, 0.f, 0.f};
    v8f acc[4];                         // O^T tiles: c-tile t, lane col = n
#pragma unroll
    for (int t = 0; t < 4; ++t) acc[t] = vzero;

    float mrun = -1e30f;                // running max (exp2 domain), this n
    float lrun = 0.f;                   // lane-local partial row sum

    // --- cooperative staging indices (256 threads) ---------------------------
    // K tile: 64x32 halves = 256 16B chunks (1/thread);
    // V tile: 64x64 halves = 512 16B chunks (2/thread).
    const int kM  = tid >> 2;           // 0..63
    const int kC8 = (tid & 3) * 8;
    const int vC  = tid >> 3;           // 0..31
    const int vM8 = (tid & 7) * 8;

#if defined(USE_ASYNC_LDS)
    auto issue_stage = [&](int buf, int m0) {
        async_b128(&Kb[((size_t)(m0 + kM)) * D + kC8], &sK[buf][kM * KLD + kC8]);
        async_b128(&Vb[(size_t)(vC)      * HW + m0 + vM8], &sV[buf][vC * VLD + vM8]);
        async_b128(&Vb[(size_t)(vC + 32) * HW + m0 + vM8], &sV[buf][(vC + 32) * VLD + vM8]);
    };
    issue_stage(0, 0);
#else
    f32x4 pre[3];
    auto issue_loads = [&](int m0) {
        pre[0] = *(const f32x4*)&Kb[((size_t)(m0 + kM)) * D + kC8];
        pre[1] = *(const f32x4*)&Vb[(size_t)(vC)      * HW + m0 + vM8];
        pre[2] = *(const f32x4*)&Vb[(size_t)(vC + 32) * HW + m0 + vM8];
    };
    issue_loads(0);
#endif

    for (int m0 = 0; m0 < HW; m0 += MT) {
#if defined(USE_ASYNC_LDS)
        const int cur = (m0 / MT) & 1;
        asm volatile("s_wait_asynccnt 0x0" ::: "memory");  // my DMAs into cur done
        __syncthreads();                                   // everyone's done
        if (m0 + MT < HW) issue_stage(cur ^ 1, m0 + MT);   // DMA next, overlap
        const _Float16* cK = sK[cur];
        const _Float16* cV = sV[cur];
#else
        __syncthreads();                 // prev-iteration readers done
        *(f32x4*)&sK[0][kM * KLD + kC8]        = pre[0];
        *(f32x4*)&sV[0][vC * VLD + vM8]        = pre[1];
        *(f32x4*)&sV[0][(vC + 32) * VLD + vM8] = pre[2];
        __syncthreads();
        if (m0 + MT < HW) issue_loads(m0 + MT);   // overlap with compute
        const _Float16* cK = sK[0];
        const _Float16* cV = sV[0];
#endif

        // S^T = K * Q^T : 4 tiles of 16 keys.  D-layout: lane holds
        // (col n = lane&15, rows m = 16j + r + 8*(lane>>4)).
        v8f s[4];
#pragma unroll
        for (int j = 0; j < 4; ++j) {
            const v16h kf = load_frag(&cK[(j * 16) * KLD], KLD, lane);
            s[j] = __builtin_amdgcn_wmma_f32_16x16x32_f16(
                false, kf, false, qf, (short)0, vzero, false, false);
        }

        // Online softmax in exp2 domain (scalar stats per lane).
        float tm = mrun;
#pragma unroll
        for (int j = 0; j < 4; ++j) {
#pragma unroll
            for (int r = 0; r < 8; ++r) tm = fmaxf(tm, s[j][r]);
        }
        tm = fmaxf(tm, __shfl_xor(tm, 16, 32));   // merge the 2 lanes sharing n
        const float alpha = __builtin_amdgcn_exp2f(mrun - tm);
        mrun = tm;

        float psum = 0.f;
#pragma unroll
        for (int j = 0; j < 4; ++j) {
#pragma unroll
            for (int r = 0; r < 8; ++r) {
                const float p = __builtin_amdgcn_exp2f(s[j][r] - tm);  // p <= 1
                s[j][r] = p;
                psum += p;
            }
        }
        lrun = lrun * alpha + psum;               // lane-local; reduce at end
#pragma unroll
        for (int t = 0; t < 4; ++t) {
#pragma unroll
            for (int r = 0; r < 8; ++r) acc[t][r] *= alpha;
        }

        // O^T += V * P^T.  P^T is the B operand: lane l, half i<8 needs
        // k = sel*8+i of m-tile (2kc)   -> that lane's VGPR i of s[2kc];
        //          i>=8 -> k-16 of tile (2kc+1) -> VGPR i-8 of s[2kc+1].
        // Packed f32->f16 conversion in registers (v_cvt_pk_rtz), no LDS.
#pragma unroll
        for (int kc = 0; kc < 2; ++kc) {
            FragU pk;
#pragma unroll
            for (int u = 0; u < 4; ++u) {
                auto lo = __builtin_amdgcn_cvt_pkrtz(s[2 * kc][2 * u],
                                                     s[2 * kc][2 * u + 1]);
                auto hi = __builtin_amdgcn_cvt_pkrtz(s[2 * kc + 1][2 * u],
                                                     s[2 * kc + 1][2 * u + 1]);
                __builtin_memcpy(&pk.e[2 * u], &lo, 4);
                __builtin_memcpy(&pk.e[8 + 2 * u], &hi, 4);
            }
#pragma unroll
            for (int t = 0; t < 4; ++t) {
                const v16h vf = load_frag(&cV[(t * 16) * VLD + kc * 32], VLD, lane);
                acc[t] = __builtin_amdgcn_wmma_f32_16x16x32_f16(
                    false, vf, false, pk.h, (short)0, acc[t], false, false);
            }
        }
    }

    // Epilogue: total row sum, then out[b][c][n] = acc^T / lsum.
    // O^T D-layout: lane holds col n = lane&15, rows c = r + 8*sel + 16t.
    lrun += __shfl_xor(lrun, 16, 32);
    const float inv = 1.f / lrun;
    const int n = n0 + (lane & 15);
    const int sel = lane >> 4;
#pragma unroll
    for (int t = 0; t < 4; ++t) {
#pragma unroll
        for (int r = 0; r < 8; ++r) {
            const int c = t * 16 + r + 8 * sel;
            out[((size_t)b * C + c) * HW + n] = acc[t][r] * inv;  // coalesced in n
        }
    }
}

// ---------------------------------------------------------------------------
extern "C" void kernel_launch(void* const* d_in, const int* in_sizes, int n_in,
                              void* d_out, int out_size, void* d_ws, size_t ws_size,
                              hipStream_t stream) {
    const float* x  = (const float*)d_in[0];
    const float* wq = (const float*)d_in[1];
    const float* wk = (const float*)d_in[2];
    const float* wv = (const float*)d_in[3];
    float* out = (float*)d_out;

    _Float16* Qh = (_Float16*)d_ws;                       // [8][4096][32]
    _Float16* Kt = Qh + (size_t)BS * HW * D;              // [8][4096][32]
    _Float16* Vh = Kt + (size_t)BS * HW * D;              // [8][64][4096]

    proj_kernel<<<BS * HW / 256, 256, 0, stream>>>(x, wq, wk, wv, Qh, Kt, Vh);
    attn_kernel<<<dim3(HW / (WAVES * 16), BS), WAVES * 32, 0, stream>>>(Qh, Kt, Vh, out);
}
